// TransformerBlock_40862318854865
// MI455X (gfx1250) — compile-verified, matrix operand order checked by reading, not run on hardware
//
#include <hip/hip_runtime.h>
#include <hip/hip_bf16.h>
#include <math.h>

// ---------- types ----------
typedef __attribute__((ext_vector_type(16))) __bf16 v16bf;
typedef __attribute__((ext_vector_type(8)))  float  v8f;
typedef __attribute__((ext_vector_type(4)))  unsigned int u32x4;

#define DEV __device__ __forceinline__

DEV unsigned short f2bf(float f) {               // f32 -> bf16 (RNE)
  unsigned int u = __float_as_uint(f);
  u += 0x7FFFu + ((u >> 16) & 1u);
  return (unsigned short)(u >> 16);
}

struct FragBF { union { v16bf v; u32x4 q[2]; }; };

// LDS byte offset of a generic pointer into a __shared__ array (AS3 ptrs are
// 32-bit LDS offsets on amdgcn).
typedef __attribute__((address_space(3))) const void* lds_cptr_t;
DEV unsigned int lds_off(const void* p) {
  return (unsigned int)(unsigned long long)(lds_cptr_t)p;
}

// CDNA5 async DMA: global -> LDS, 16B per lane, tracked by ASYNCcnt.
DEV void async_b128_to_lds(unsigned int lds_addr, const void* gaddr) {
  asm volatile("global_load_async_to_lds_b128 %0, %1, off"
               :: "v"(lds_addr), "v"((unsigned long long)gaddr) : "memory");
}
DEV void wait_async0() {
  asm volatile("s_wait_asynccnt 0x0" ::: "memory");
}

// A/B fragment fetch from an LDS row (row base 16B-aligned, stride 40 u16 = 80B).
// CDNA5 16-bit operand layout: lane%16 selects the row/col, lane/16 the K-half;
// elements 0..7 -> K = kh*8.., elements 8..15 -> K = 16+kh*8.. .
DEV FragBF ld_frag_lds(const unsigned short* rowbase, int kh) {
  FragBF f;
  f.q[0] = *(const u32x4*)(rowbase + kh * 8);
  f.q[1] = *(const u32x4*)(rowbase + 16 + kh * 8);
  return f;
}

DEV v8f vzero8() { v8f z = {0.f,0.f,0.f,0.f,0.f,0.f,0.f,0.f}; return z; }

DEV v8f wmma_bf16(const FragBF& a, const FragBF& b, v8f c) {
  return __builtin_amdgcn_wmma_f32_16x16x32_bf16(
      false, a.v, false, b.v, (short)0, c, false, false);
}

// ---------- fused cast + transpose: in[K][N] f32 -> out[N][K] bf16 ----------
__global__ __launch_bounds__(256) void castT_bf16_kernel(
    const float* __restrict__ in, unsigned short* __restrict__ out,
    int K, int N) {
  __shared__ float tile[32][33];
  const int k0 = blockIdx.y * 32, n0 = blockIdx.x * 32;
  const int tx = threadIdx.x & 31, ty = threadIdx.x >> 5;   // 32 x 8
#pragma unroll
  for (int r = 0; r < 4; ++r)
    tile[ty + r * 8][tx] = in[(size_t)(k0 + ty + r * 8) * N + n0 + tx];
  __syncthreads();
#pragma unroll
  for (int r = 0; r < 4; ++r)
    out[(size_t)(n0 + ty + r * 8) * K + k0 + tx] = f2bf(tile[tx][ty + r * 8]);
}

// ---------- LayerNorm (D = 1024), bf16 output ----------
__global__ __launch_bounds__(256) void ln_bf16_kernel(
    const float* __restrict__ x, const float* __restrict__ g,
    const float* __restrict__ bt, unsigned short* __restrict__ out) {
  __shared__ float red[256];
  const int row = blockIdx.x, tid = threadIdx.x;
  const float* xr = x + (size_t)row * 1024;
  float4 f = ((const float4*)xr)[tid];
  red[tid] = f.x + f.y + f.z + f.w;
  __syncthreads();
  for (int off = 128; off > 0; off >>= 1) {
    if (tid < off) red[tid] += red[tid + off];
    __syncthreads();
  }
  const float mu = red[0] * (1.0f / 1024.0f);
  __syncthreads();
  const float dx = f.x - mu, dy = f.y - mu, dz = f.z - mu, dw = f.w - mu;
  red[tid] = dx * dx + dy * dy + dz * dz + dw * dw;
  __syncthreads();
  for (int off = 128; off > 0; off >>= 1) {
    if (tid < off) red[tid] += red[tid + off];
    __syncthreads();
  }
  const float inv = rsqrtf(red[0] * (1.0f / 1024.0f) + 1e-5f);
  const int c = tid * 4;
  unsigned short* orow = out + (size_t)row * 1024 + c;
  orow[0] = f2bf(dx * inv * g[c + 0] + bt[c + 0]);
  orow[1] = f2bf(dy * inv * g[c + 1] + bt[c + 1]);
  orow[2] = f2bf(dz * inv * g[c + 2] + bt[c + 2]);
  orow[3] = f2bf(dw * inv * g[c + 3] + bt[c + 3]);
}

// ---------- bf16 WMMA GEMM: C[M,N] = A[M,K] @ Bt[N,K]^T (+bias, gelu, resid) ----
// 128x128 block tile, 8 waves x (32x64). Double-buffered LDS fed by
// global_load_async_to_lds_b128 (ASYNCcnt): one barrier per K tile, DMA
// overlapped with the WMMA pipe. All 12 fragment ds_loads are issued before
// the 8 WMMAs so the matrix pipe runs back-to-back after one dscnt wait.
__global__ __launch_bounds__(256) void gemm_bf16_kernel(
    const unsigned short* __restrict__ A, const unsigned short* __restrict__ Bt,
    const float* __restrict__ bias, const float* __restrict__ resid,
    float* __restrict__ Cf, unsigned short* __restrict__ Cbf,
    int M, int N, int K, int dogelu) {
  __shared__ alignas(16) unsigned short sA[2][128 * 40];  // [m][k], k-pad to 40
  __shared__ alignas(16) unsigned short sB[2][128 * 40];  // [n][k]
  const int tid = threadIdx.x;
  const int w = tid >> 5, lane = tid & 31, ln = lane & 15, kh = lane >> 4;
  const int wrow = (w & 3) * 32, wcol = (w >> 2) * 64;
  const int mrow0 = blockIdx.y * 128, ncol0 = blockIdx.x * 128;
  const unsigned short* Ab = A + (size_t)mrow0 * K;     // tile row 0
  const unsigned short* Bb = Bt + (size_t)ncol0 * K;

  // this thread's two 16B chunks per 128x32 tile: i = tid, tid+256
  const int r0 = tid >> 2, s0 = (tid & 3) * 8;
  const int r1 = (tid + 256) >> 2, s1 = ((tid + 256) & 3) * 8;

  v8f c[2][4];
#pragma unroll
  for (int mi = 0; mi < 2; ++mi)
#pragma unroll
    for (int ni = 0; ni < 4; ++ni) c[mi][ni] = vzero8();

  const int nk = K >> 5;
  // prologue: DMA tile 0 into buffer 0
  async_b128_to_lds(lds_off(&sA[0][r0 * 40 + s0]), Ab + (size_t)r0 * K + s0);
  async_b128_to_lds(lds_off(&sA[0][r1 * 40 + s1]), Ab + (size_t)r1 * K + s1);
  async_b128_to_lds(lds_off(&sB[0][r0 * 40 + s0]), Bb + (size_t)r0 * K + s0);
  async_b128_to_lds(lds_off(&sB[0][r1 * 40 + s1]), Bb + (size_t)r1 * K + s1);
  wait_async0();
  __syncthreads();

  for (int kt = 0; kt < nk; ++kt) {
    const int cur = kt & 1, nxt = cur ^ 1;
    if (kt + 1 < nk) {      // DMA next tile while this tile runs on the WMMA pipe
      const int k1 = (kt + 1) * 32;
      async_b128_to_lds(lds_off(&sA[nxt][r0 * 40 + s0]), Ab + (size_t)r0 * K + k1 + s0);
      async_b128_to_lds(lds_off(&sA[nxt][r1 * 40 + s1]), Ab + (size_t)r1 * K + k1 + s1);
      async_b128_to_lds(lds_off(&sB[nxt][r0 * 40 + s0]), Bb + (size_t)r0 * K + k1 + s0);
      async_b128_to_lds(lds_off(&sB[nxt][r1 * 40 + s1]), Bb + (size_t)r1 * K + k1 + s1);
    }
    if (kt + 2 < nk) {      // warm GL2 for the tile after next
      __builtin_prefetch(Ab + (size_t)r0 * K + (kt + 2) * 32, 0, 1);
      __builtin_prefetch(Bb + (size_t)r0 * K + (kt + 2) * 32, 0, 1);
    }

    // issue every fragment load first, then run the 8 WMMAs back-to-back
    FragBF af[2], bf[4];
#pragma unroll
    for (int mi = 0; mi < 2; ++mi)
      af[mi] = ld_frag_lds(&sA[cur][(wrow + mi * 16 + ln) * 40], kh);
#pragma unroll
    for (int ni = 0; ni < 4; ++ni)
      bf[ni] = ld_frag_lds(&sB[cur][(wcol + ni * 16 + ln) * 40], kh);
#pragma unroll
    for (int ni = 0; ni < 4; ++ni) {
      c[0][ni] = wmma_bf16(af[0], bf[ni], c[0][ni]);
      c[1][ni] = wmma_bf16(af[1], bf[ni], c[1][ni]);
    }

    wait_async0();          // own DMA done writing LDS
    __syncthreads();        // everyone's DMA done + all reads of `cur` finished
  }

  // epilogue — C layout: M = r + 8*(lane>=16), N = lane%16
#pragma unroll
  for (int mi = 0; mi < 2; ++mi) {
#pragma unroll
    for (int ni = 0; ni < 4; ++ni) {
      const int col = ncol0 + wcol + ni * 16 + ln;
      const float bv = bias ? bias[col] : 0.0f;
#pragma unroll
      for (int r = 0; r < 8; ++r) {
        const int row = mrow0 + wrow + mi * 16 + r + 8 * kh;
        float v = c[mi][ni][r] + bv;
        if (dogelu) v = 0.5f * v * (1.0f + erff(v * 0.70710678118f));
        if (resid) v += resid[(size_t)row * N + col];
        if (Cf)  Cf[(size_t)row * N + col] = v;
        if (Cbf) Cbf[(size_t)row * N + col] = f2bf(v);
      }
    }
  }
}

// ---------- causal flash attention over bf16 qkv [4096, 3072] ----------
// grid (B*H = 32, T/128 = 16); each wave owns 16 query rows; keys in chunks of 32.
__global__ __launch_bounds__(256) void attn_kernel(
    const unsigned short* __restrict__ qkv, unsigned short* __restrict__ outbf) {
  __shared__ alignas(16) unsigned short Kl[32 * 72];        // K chunk [key][dim]
  __shared__ alignas(16) unsigned short Vt[64 * 40];        // V chunk [dim][key]
  __shared__ alignas(16) unsigned short Pl[8 * 16 * 40];    // per-wave P staging
  const int tid = threadIdx.x;
  const int w = tid >> 5, lane = tid & 31, ln = lane & 15, kh = lane >> 4;
  const int b = blockIdx.x >> 4, h = blockIdx.x & 15;
  const int q0 = blockIdx.y * 128;
  const int qr = q0 + w * 16;
  const float scale = 0.125f;                               // 1/sqrt(64)
  const size_t rs = 3072;
  const unsigned short* base = qkv + (size_t)b * 2048 * rs + h * 64;

  // Q fragments (16 rows x 64 dims = 2 k-steps) straight from global
  FragBF qf[2];
  {
    const unsigned short* qrow = base + (size_t)(qr + ln) * rs;
    qf[0].q[0] = *(const u32x4*)(qrow + kh * 8);
    qf[0].q[1] = *(const u32x4*)(qrow + 16 + kh * 8);
    qf[1].q[0] = *(const u32x4*)(qrow + 32 + kh * 8);
    qf[1].q[1] = *(const u32x4*)(qrow + 48 + kh * 8);
  }

  v8f o[4];
  float m[8], l[8];
#pragma unroll
  for (int nt = 0; nt < 4; ++nt) o[nt] = vzero8();
#pragma unroll
  for (int r = 0; r < 8; ++r) { m[r] = -INFINITY; l[r] = 0.0f; }

  const int nchunk = (q0 + 128) >> 5;
  for (int jc = 0; jc < nchunk; ++jc) {
    const int kb0 = jc << 5;
    __syncthreads();
    {   // K via async DMA to LDS; V register-staged (needs transpose)
      const int idx = tid * 8;
      const int key = idx >> 6, d0 = idx & 63;
      const unsigned short* srck = base + (size_t)(kb0 + key) * rs + 1024 + d0;
      async_b128_to_lds(lds_off(&Kl[key * 72 + d0]), srck);
      const unsigned short* srcv = base + (size_t)(kb0 + key) * rs + 2048 + d0;
      union { u32x4 q; unsigned short s[8]; } t;
      t.q = *(const u32x4*)srcv;
#pragma unroll
      for (int e = 0; e < 8; ++e) Vt[(d0 + e) * 40 + key] = t.s[e];
    }
    wait_async0();
    __syncthreads();
    if (kb0 >= qr + 16) continue;   // fully masked chunk for this wave

    // S = Q K^T for 2 key sub-tiles of 16 (fragment: col = key, contiguous dims)
    float a0[8], a1[8];
    {
      FragBF kf[4];   // [js][kstep] preloaded, then 4 WMMAs back-to-back
      kf[0].q[0] = *(const u32x4*)&Kl[ln * 72 + kh * 8];
      kf[0].q[1] = *(const u32x4*)&Kl[ln * 72 + 16 + kh * 8];
      kf[1].q[0] = *(const u32x4*)&Kl[ln * 72 + 32 + kh * 8];
      kf[1].q[1] = *(const u32x4*)&Kl[ln * 72 + 48 + kh * 8];
      kf[2].q[0] = *(const u32x4*)&Kl[(16 + ln) * 72 + kh * 8];
      kf[2].q[1] = *(const u32x4*)&Kl[(16 + ln) * 72 + 16 + kh * 8];
      kf[3].q[0] = *(const u32x4*)&Kl[(16 + ln) * 72 + 32 + kh * 8];
      kf[3].q[1] = *(const u32x4*)&Kl[(16 + ln) * 72 + 48 + kh * 8];
      v8f s0v = vzero8(), s1v = vzero8();
      s0v = wmma_bf16(qf[0], kf[0], s0v);
      s0v = wmma_bf16(qf[1], kf[1], s0v);
      s1v = wmma_bf16(qf[0], kf[2], s1v);
      s1v = wmma_bf16(qf[1], kf[3], s1v);
#pragma unroll
      for (int r = 0; r < 8; ++r) { a0[r] = s0v[r]; a1[r] = s1v[r]; }
    }

    // online softmax; row reductions across each 16-lane half via shfl_xor
    float alp[8], p0[8], p1[8];
#pragma unroll
    for (int r = 0; r < 8; ++r) {
      const int row = qr + r + 8 * kh;
      float x0 = (kb0 + ln      <= row) ? a0[r] * scale : -INFINITY;
      float x1 = (kb0 + 16 + ln <= row) ? a1[r] * scale : -INFINITY;
      float vm = fmaxf(x0, x1);
      vm = fmaxf(vm, __shfl_xor(vm, 1, 32));
      vm = fmaxf(vm, __shfl_xor(vm, 2, 32));
      vm = fmaxf(vm, __shfl_xor(vm, 4, 32));
      vm = fmaxf(vm, __shfl_xor(vm, 8, 32));
      const float mnew  = fmaxf(m[r], vm);
      const float msafe = (mnew == -INFINITY) ? 0.0f : mnew;
      alp[r] = (m[r] == -INFINITY) ? 0.0f : __expf(m[r] - mnew);
      p0[r] = __expf(x0 - msafe);
      p1[r] = __expf(x1 - msafe);
      float rsum = p0[r] + p1[r];
      rsum += __shfl_xor(rsum, 1, 32);
      rsum += __shfl_xor(rsum, 2, 32);
      rsum += __shfl_xor(rsum, 4, 32);
      rsum += __shfl_xor(rsum, 8, 32);
      l[r] = l[r] * alp[r] + rsum;
      m[r] = mnew;
    }
#pragma unroll
    for (int nt = 0; nt < 4; ++nt)
#pragma unroll
      for (int r = 0; r < 8; ++r) o[nt][r] *= alp[r];

    // stage P (16x32) through wave-private LDS into A-fragment layout
    unsigned short* pw = &Pl[w * 16 * 40];
#pragma unroll
    for (int r = 0; r < 8; ++r) {
      pw[(r + 8 * kh) * 40 + ln]      = f2bf(p0[r]);
      pw[(r + 8 * kh) * 40 + 16 + ln] = f2bf(p1[r]);
    }
    FragBF pf, vf[4];
    pf.q[0] = *(const u32x4*)&pw[ln * 40 + kh * 8];
    pf.q[1] = *(const u32x4*)&pw[ln * 40 + 16 + kh * 8];
#pragma unroll
    for (int nt = 0; nt < 4; ++nt) {   // V frag: col = dim, contiguous keys
      vf[nt].q[0] = *(const u32x4*)&Vt[(nt * 16 + ln) * 40 + kh * 8];
      vf[nt].q[1] = *(const u32x4*)&Vt[(nt * 16 + ln) * 40 + 16 + kh * 8];
    }
#pragma unroll
    for (int nt = 0; nt < 4; ++nt) o[nt] = wmma_bf16(pf, vf[nt], o[nt]);
  }

  // normalize + store bf16 [B*T, 1024]
#pragma unroll
  for (int r = 0; r < 8; ++r) {
    const float li = l[r];
    const float invl = (li > 0.0f) ? 1.0f / li : 0.0f;
    const int row = qr + r + 8 * kh;
    unsigned short* orow = outbf + (size_t)(b * 2048 + row) * 1024 + h * 64;
#pragma unroll
    for (int nt = 0; nt < 4; ++nt) orow[nt * 16 + ln] = f2bf(o[nt][r] * invl);
  }
}

// ---------- launch ----------
extern "C" void kernel_launch(void* const* d_in, const int* in_sizes, int n_in,
                              void* d_out, int out_size, void* d_ws, size_t ws_size,
                              hipStream_t stream) {
  (void)in_sizes; (void)n_in; (void)out_size; (void)ws_size;
  const float* x    = (const float*)d_in[0];
  const float* ln1g = (const float*)d_in[1];
  const float* ln1b = (const float*)d_in[2];
  const float* Wqkv = (const float*)d_in[3];
  const float* bqkv = (const float*)d_in[4];
  const float* Wout = (const float*)d_in[5];
  const float* bout = (const float*)d_in[6];
  const float* ln2g = (const float*)d_in[7];
  const float* ln2b = (const float*)d_in[8];
  const float* W1   = (const float*)d_in[9];
  const float* b1   = (const float*)d_in[10];
  const float* W2   = (const float*)d_in[11];
  const float* b2   = (const float*)d_in[12];

  const int M = 4096, D = 1024, HID = 4096, N3 = 3072;
  char* ws = (char*)d_ws;
  const size_t MB = 1ull << 20;
  // lifetime-based workspace reuse (94 MB peak); weights stored transposed [N][K]
  unsigned short* hbf   = (unsigned short*)(ws + 0);        // 8MB  (region A)
  unsigned short* wqkvb = (unsigned short*)(ws + 8 * MB);   // 6MB  (region A)
  unsigned short* attnb = (unsigned short*)(ws + 0);        // 8MB  (reuse A after QKV)
  unsigned short* woutb = (unsigned short*)(ws + 8 * MB);   // 2MB  (reuse A after QKV)
  unsigned short* qkvb  = (unsigned short*)(ws + 14 * MB);  // 24MB (region B)
  unsigned short* h2b   = (unsigned short*)(ws + 14 * MB);  // 8MB  (reuse B after attn)
  unsigned short* w1b   = (unsigned short*)(ws + 22 * MB);  // 8MB  (reuse B after attn)
  float*          x1    = (float*)(ws + 38 * MB);           // 16MB
  unsigned short* hidb  = (unsigned short*)(ws + 54 * MB);  // 32MB
  unsigned short* w2b   = (unsigned short*)(ws + 86 * MB);  // 8MB

  // pre-LN attention
  ln_bf16_kernel<<<M, 256, 0, stream>>>(x, ln1g, ln1b, hbf);
  castT_bf16_kernel<<<dim3(N3 / 32, D / 32), 256, 0, stream>>>(Wqkv, wqkvb, D, N3);
  gemm_bf16_kernel<<<dim3(N3 / 128, M / 128), 256, 0, stream>>>(
      hbf, wqkvb, bqkv, nullptr, nullptr, qkvb, M, N3, D, 0);
  attn_kernel<<<dim3(32, 16), 256, 0, stream>>>(qkvb, attnb);
  castT_bf16_kernel<<<dim3(D / 32, D / 32), 256, 0, stream>>>(Wout, woutb, D, D);
  gemm_bf16_kernel<<<dim3(D / 128, M / 128), 256, 0, stream>>>(
      attnb, woutb, bout, x, x1, nullptr, M, D, D, 0);

  // pre-LN MLP
  ln_bf16_kernel<<<M, 256, 0, stream>>>(x1, ln2g, ln2b, h2b);
  castT_bf16_kernel<<<dim3(HID / 32, D / 32), 256, 0, stream>>>(W1, w1b, D, HID);
  gemm_bf16_kernel<<<dim3(HID / 128, M / 128), 256, 0, stream>>>(
      h2b, w1b, b1, nullptr, nullptr, hidb, M, HID, D, 1);
  castT_bf16_kernel<<<dim3(D / 32, HID / 32), 256, 0, stream>>>(W2, w2b, HID, D);
  gemm_bf16_kernel<<<dim3(D / 128, M / 128), 256, 0, stream>>>(
      hidb, w2b, b2, x1, (float*)d_out, nullptr, M, D, HID, 0);
}
